// Router_33225867002144
// MI455X (gfx1250) — matrix operand order repaired and checked by reading, not compiled, not run on hardware
//
#include <hip/hip_runtime.h>

// Problem constants
#define BT   16384   // B*T
#define T_   2048
#define B_   8
#define D_   1024
#define E_   8
#define F_   4096

typedef __bf16 bf16_t;
typedef __attribute__((ext_vector_type(4)))  __bf16 bf16x4;
typedef __attribute__((ext_vector_type(8)))  __bf16 bf16x8;
typedef __attribute__((ext_vector_type(16))) __bf16 v16bf;
typedef __attribute__((ext_vector_type(8)))  float  v8f;

__device__ __forceinline__ bf16_t f2bf(float f) {
  unsigned u = __builtin_bit_cast(unsigned, f);
  unsigned r = (u + 0x7FFFu + ((u >> 16) & 1u)) >> 16;
  unsigned short s = (unsigned short)r;
  return __builtin_bit_cast(bf16_t, s);
}

// A-fragment (16x32 bf16): lane holds row M=l%16; K = kb..kb+7 (elems 0-7) and
// kb+16..kb+23 (elems 8-15), kb = (lane<16)?0:8. p points at row + d0 + kb.
__device__ __forceinline__ v16bf load_fragA(const bf16_t* p) {
  bf16x8 lo = *(const bf16x8*)p;
  bf16x8 hi = *(const bf16x8*)(p + 16);
  return __builtin_shufflevector(lo, hi, 0,1,2,3,4,5,6,7,8,9,10,11,12,13,14,15);
}
// B-fragment (32x16 bf16): lane holds col N=l%16; K = kb..kb+7 (elems 0-7) and
// kb+16..kb+23 (elems 8-15), kb = (lane<16)?0:8. p points at row(N) + k0 + kb.
__device__ __forceinline__ v16bf load_fragB(const bf16_t* p) {
  bf16x8 lo = *(const bf16x8*)p;
  bf16x8 hi = *(const bf16x8*)(p + 16);
  return __builtin_shufflevector(lo, hi, 0,1,2,3,4,5,6,7,8,9,10,11,12,13,14,15);
}

// ---------------- prep kernels ----------------

__global__ void cvt_f32_to_bf16(const float* __restrict__ s, bf16_t* __restrict__ d, int n4) {
  int i = blockIdx.x * blockDim.x + threadIdx.x;
  if (i < n4) {
    float4 v = ((const float4*)s)[i];
    bf16x4 o;
    o[0] = f2bf(v.x); o[1] = f2bf(v.y); o[2] = f2bf(v.z); o[3] = f2bf(v.w);
    ((bf16x4*)d)[i] = o;
  }
}

__global__ void zero_f32(float4* __restrict__ p, int n4) {
  int i = blockIdx.x * blockDim.x + threadIdx.x;
  if (i < n4) p[i] = make_float4(0.f, 0.f, 0.f, 0.f);
}

// gate logits + top-2 per token; one wave per token (wave32)
__global__ void gate_top2_kernel(const float* __restrict__ tokens,
                                 const float* __restrict__ gw,
                                 const float* __restrict__ gb,
                                 float* __restrict__ v0, float* __restrict__ v1,
                                 int* __restrict__ i0, int* __restrict__ i1) {
  __shared__ float gws[E_ * D_];
  for (int i = threadIdx.x; i < E_ * D_; i += blockDim.x) gws[i] = gw[i];
  __syncthreads();
  int wave = threadIdx.x >> 5, lane = threadIdx.x & 31;
  int t = blockIdx.x * (blockDim.x >> 5) + wave;
  const float* xr = tokens + (size_t)t * D_;
  float acc[E_];
#pragma unroll
  for (int e = 0; e < E_; ++e) acc[e] = 0.f;
  for (int j = 0; j < D_ / 32; ++j) {
    float x = xr[lane + 32 * j];
#pragma unroll
    for (int e = 0; e < E_; ++e) acc[e] += x * gws[e * D_ + lane + 32 * j];
  }
#pragma unroll
  for (int e = 0; e < E_; ++e) {
#pragma unroll
    for (int off = 16; off; off >>= 1) acc[e] += __shfl_xor(acc[e], off, 32);
    acc[e] += gb[e];
  }
  if (lane == 0) {
    float b = -1e30f, s = -1e30f; int bi = 0, si = 0;
#pragma unroll
    for (int e = 0; e < E_; ++e) {
      float v = acc[e];
      if (v > b)      { s = b; si = bi; b = v; bi = e; }
      else if (v > s) { s = v; si = e; }
    }
    v0[t] = b; i0[t] = bi; v1[t] = s; i1[t] = si;
  }
}

// softmax over the TOKEN axis (axis=1) per (batch, slot), as in the reference
__global__ void softmax_T_kernel(float* __restrict__ v0, float* __restrict__ v1) {
  int b = blockIdx.x >> 1, s = blockIdx.x & 1;
  float* p = (s ? v1 : v0) + b * T_;
  __shared__ float red[256];
  int tid = threadIdx.x;
  float m = -1e30f;
  for (int j = tid; j < T_; j += 256) m = fmaxf(m, p[j]);
  red[tid] = m; __syncthreads();
  for (int off = 128; off; off >>= 1) { if (tid < off) red[tid] = fmaxf(red[tid], red[tid + off]); __syncthreads(); }
  float mx = red[0]; __syncthreads();
  float sum = 0.f;
  for (int j = tid; j < T_; j += 256) sum += __expf(p[j] - mx);
  red[tid] = sum; __syncthreads();
  for (int off = 128; off; off >>= 1) { if (tid < off) red[tid] += red[tid + off]; __syncthreads(); }
  float inv = 1.f / red[0];
  for (int j = tid; j < T_; j += 256) p[j] = __expf(p[j] - mx) * inv;
}

// deterministic per-expert compaction (block-wide prefix scan, no atomics)
__global__ void compact_kernel(const int* __restrict__ i0, const int* __restrict__ i1,
                               const float* __restrict__ w0, const float* __restrict__ w1,
                               int* __restrict__ ltok, float* __restrict__ lw,
                               int* __restrict__ counts) {
  int e = blockIdx.x;
  int* lt = ltok + e * BT;
  float* lww = lw + e * BT;
  __shared__ int sc[256];
  __shared__ int sbase;
  int tid = threadIdx.x;
  if (tid == 0) sbase = 0;
  __syncthreads();
  for (int c = 0; c < BT / 256; ++c) {
    int base = sbase;
    int t = c * 256 + tid;
    int a = i0[t], b = i1[t];
    int f = (a == e) || (b == e);
    float w = (a == e) ? w0[t] : ((b == e) ? w1[t] : 0.f);
    sc[tid] = f; __syncthreads();
    for (int off = 1; off < 256; off <<= 1) {
      int v = (tid >= off) ? sc[tid - off] : 0;
      __syncthreads();
      sc[tid] += v;
      __syncthreads();
    }
    int incl = sc[tid];
    int total = sc[255];
    if (f) { lt[base + incl - 1] = t; lww[base + incl - 1] = w; }
    __syncthreads();
    if (tid == 0) sbase = base + total;
    __syncthreads();
  }
  int cnt = sbase;
  if (tid == 0) counts[e] = cnt;
  int rounded = (cnt + 63) & ~63;
  for (int p = cnt + tid; p < rounded; p += 256) { lt[p] = 0; lww[p] = 0.f; }
}

// ---------------- fused two-layer expert GEMM ----------------
// 512 threads = 16 waves. M-tile = 64 gathered tokens.
// y[64x1024] accumulated in VGPRs: wave w owns output cols [w*64, w*64+64).
#define XS_STR 1032   // 1024 + 8 halves pad
#define W2_STR 72     // 64 + 8 halves pad
#define H_STR  72

__global__ __launch_bounds__(512)
void moe_expert_kernel(const bf16_t* __restrict__ tokb,
                       const bf16_t* __restrict__ w1b,
                       const bf16_t* __restrict__ w2b,
                       const float* __restrict__ b1,
                       const float* __restrict__ b2,
                       const int* __restrict__ ltok,
                       const float* __restrict__ lw,
                       const int* __restrict__ counts,
                       int e, float* __restrict__ out) {
  const int cnt = counts[e];
  const int m0 = blockIdx.x * 64;
  if (m0 >= cnt) return;

  const bf16_t* w1e = w1b + (size_t)e * F_ * D_;
  const bf16_t* w2e = w2b + (size_t)e * D_ * F_;
  const float* b1e = b1 + e * F_;
  const float* b2e = b2 + e * D_;
  const int* lte = ltok + e * BT + m0;
  const float* lwe = lw + e * BT + m0;

  extern __shared__ char smem[];
  bf16_t* Xs = (bf16_t*)smem;              // 64 * XS_STR halves  (129 KB)
  bf16_t* Wb = Xs + 64 * XS_STR;           // max(64*XS_STR, 1024*W2_STR) = 73728 halves (144 KB)
  bf16_t* Hs = Wb + 1024 * W2_STR;         // 64 * H_STR halves   (9 KB)
  float* b1c = (float*)(Hs + 64 * H_STR);  // 64
  float* gs  = b1c + 64;                   // 64
  int*   ts  = (int*)(gs + 64);            // 64

  const int tid  = threadIdx.x;
  const int wv   = tid >> 5;
  const int lane = tid & 31;
  const int half = lane >> 4;
  const int ln   = lane & 15;
  const int mi   = wv >> 2, fi = wv & 3;
  const int kbA  = half * 8;
  const int kbB  = half * 8;

  if (tid < 64) { ts[tid] = lte[tid]; gs[tid] = lwe[tid]; }
  __syncthreads();

  // stage gathered X tile [64 x 1024] bf16 into LDS (resident for whole kernel)
  {
    int row = tid >> 3, part = tid & 7;
    const bf16_t* src = tokb + (size_t)ts[row] * D_;
    bf16_t* dst = Xs + row * XS_STR;
#pragma unroll
    for (int c = 0; c < 16; ++c) {
      int off = (c * 8 + part) * 8;
      *(uint4*)(dst + off) = *(const uint4*)(src + off);
    }
  }

  v8f acc[4][4];
#pragma unroll
  for (int i = 0; i < 4; ++i)
#pragma unroll
    for (int j = 0; j < 4; ++j)
#pragma unroll
      for (int v = 0; v < 8; ++v) acc[i][j][v] = 0.f;

  for (int fc = 0; fc < F_ / 64; ++fc) {
    __syncthreads();  // previous phase-2 finished with Wb/Hs

    // stage W1 chunk [64 f x 1024 d] + bias chunk
    {
      int row = tid >> 3, part = tid & 7;
      const bf16_t* src = w1e + (size_t)(fc * 64 + row) * D_;
      bf16_t* dst = Wb + row * XS_STR;
#pragma unroll
      for (int c = 0; c < 16; ++c) {
        int off = (c * 8 + part) * 8;
        *(uint4*)(dst + off) = *(const uint4*)(src + off);
      }
      if (tid < 64) b1c[tid] = b1e[fc * 64 + tid];
    }
    __syncthreads();

    // phase 1: h tile (mi, fi) = X[16m x 1024] @ W1chunk[16f x 1024]^T
    v8f hacc;
#pragma unroll
    for (int v = 0; v < 8; ++v) hacc[v] = 0.f;
    const bf16_t* xrow = Xs + (mi * 16 + ln) * XS_STR + kbA;
    const bf16_t* wrow = Wb + (fi * 16 + ln) * XS_STR + kbB;
#pragma unroll 4
    for (int d0 = 0; d0 < D_; d0 += 32) {
      v16bf a = load_fragA(xrow + d0);
      v16bf b = load_fragB(wrow + d0);
      hacc = __builtin_amdgcn_wmma_f32_16x16x32_bf16(false, a, false, b,
                                                     (short)0, hacc, false, false);
    }
    {
      int fcol = fi * 16 + ln;
      float bias = b1c[fcol];
#pragma unroll
      for (int v = 0; v < 8; ++v) {
        float h = hacc[v] + bias;
        h = h > 0.f ? h : 0.f;
        Hs[(mi * 16 + half * 8 + v) * H_STR + fcol] = f2bf(h);
      }
    }
    __syncthreads();

    // stage W2 chunk f 0..31 only [1024 n x 32 f] into the same buffer
#pragma unroll
    for (int i = 0; i < 8; ++i) {
      int task = tid + i * 512;
      int n = task >> 2, ch = task & 3;
      *(uint4*)(Wb + n * W2_STR + ch * 8) =
          *(const uint4*)(w2e + (size_t)n * F_ + fc * 64 + ch * 8);
    }
    // prefetch next W1 chunk while phase-2 runs
    if (fc + 1 < F_ / 64) {
      const char* pf = (const char*)(w1e + (size_t)(fc + 1) * 64 * D_);
      __builtin_prefetch(pf + tid * 256, 0, 1);
    }
    __syncthreads();

    // phase 2: y[64 x (wave's 64 cols)] += h[64x64] @ W2chunk^T
#pragma unroll
    for (int kk = 0; kk < 64; kk += 32) {
      v16bf af[4], bfr[4];
#pragma unroll
      for (int i = 0; i < 4; ++i)
        af[i] = load_fragA(Hs + (i * 16 + ln) * H_STR + kk + kbA);
#pragma unroll
      for (int j = 0; j < 4; ++j)
        bfr[j] = load_fragB(Wb + (wv * 64 + j * 16 + ln) * W2_STR + kk + kbB);
#pragma unroll
      for (int i = 0; i < 4; ++i)
#pragma unroll
        for (int j = 0; j < 4; ++j)
          acc[i][j] = __builtin_amdgcn_wmma_f32_16x16x32_bf16(false, af[i], false, bfr[j],
                                                              (short)0, acc[i][j], false, false);
    }
  }

  // epilogue: out[token, n] += gate * (y + b2[n]); unique writer per token here
#pragma unroll
  for (int j = 0; j < 4; ++j) {
    int n = wv * 64 + j * 16 + ln;
    float b2n = b2e[n];
#pragma unroll
    for (int i = 0; i < 4; ++i) {
#pragma unroll
      for (int v = 0; v < 8; ++v) {
        int ml = i * 16 + half * 8 + v;
        if (m0 + ml < cnt) {
          float* po = out + (size_t)ts[ml] * D_ + n;
          *po += gs[ml] * (acc[i][j][v] + b2n);
        }
      }
    }
  }
}

// ---------------- host ----------------

extern "C" void kernel_launch(void* const* d_in, const int* in_sizes, int n_in,
                              void* d_out, int out_size, void* d_ws, size_t ws_size,
                              hipStream_t stream) {
  (void)in_sizes; (void)n_in; (void)out_size; (void)ws_size;
  const float* tokens = (const float*)d_in[0];
  const float* gate_w = (const float*)d_in[1];
  const float* gate_b = (const float*)d_in[2];
  const float* w1     = (const float*)d_in[3];
  const float* b1     = (const float*)d_in[4];
  const float* w2     = (const float*)d_in[5];
  const float* b2     = (const float*)d_in[6];
  float* out = (float*)d_out;

  char* ws = (char*)d_ws;
  bf16_t* tokb = (bf16_t*)(ws);                       // 32 MB
  bf16_t* w1b  = (bf16_t*)(ws + 33554432ull);         // 64 MB
  bf16_t* w2b  = (bf16_t*)(ws + 100663296ull);        // 64 MB
  float*  v0   = (float*)(ws + 167772160ull);
  float*  v1   = (float*)(ws + 167837696ull);
  int*    i0   = (int*)(ws + 167903232ull);
  int*    i1   = (int*)(ws + 167968768ull);
  int*    ltok = (int*)(ws + 168034304ull);
  float*  lwv  = (float*)(ws + 168558592ull);
  int*    cnts = (int*)(ws + 169082880ull);

  cvt_f32_to_bf16<<<(BT * D_ / 4 + 255) / 256, 256, 0, stream>>>(tokens, tokb, BT * D_ / 4);
  cvt_f32_to_bf16<<<(E_ * F_ * D_ / 4 + 255) / 256, 256, 0, stream>>>(w1, w1b, E_ * F_ * D_ / 4);
  cvt_f32_to_bf16<<<(E_ * D_ * F_ / 4 + 255) / 256, 256, 0, stream>>>(w2, w2b, E_ * D_ * F_ / 4);
  zero_f32<<<(BT * D_ / 4 + 255) / 256, 256, 0, stream>>>((float4*)out, BT * D_ / 4);

  gate_top2_kernel<<<BT / 8, 256, 0, stream>>>(tokens, gate_w, gate_b, v0, v1, i0, i1);
  softmax_T_kernel<<<B_ * 2, 256, 0, stream>>>(v0, v1);
  compact_kernel<<<E_, 256, 0, stream>>>(i0, i1, v0, v1, ltok, lwv, cnts);

  size_t shmem = (size_t)(64 * XS_STR + 1024 * W2_STR + 64 * H_STR) * 2 + 3 * 64 * 4;  // 289,536 B
  hipFuncSetAttribute((const void*)moe_expert_kernel,
                      hipFuncAttributeMaxDynamicSharedMemorySize, (int)shmem);
  for (int e = 0; e < E_; ++e) {
    // experts sequential on the stream -> deterministic read-modify-write of out
    moe_expert_kernel<<<BT / 64, 512, shmem, stream>>>(tokb, w1b, w2b, b1, b2,
                                                       ltok, lwv, cnts, e, out);
  }
}
